// Encoder_3556232921377
// MI455X (gfx1250) — compile-verified, matrix operand order checked by reading, not run on hardware
//
#include <hip/hip_runtime.h>

// ---------------------------------------------------------------------------
// Bidirectional Mamba-v2 stack for MI455X (gfx1250, wave32, WMMA + TDM).
// fp32 in/out; GEMMs run on v_wmma_f32_16x16x32_bf16. GEMM weight tiles are
// DMA'd into LDS by the Tensor Data Mover (tensor_load_to_lds +
// s_wait_tensorcnt), converted fp32->bf16 in LDS once per block, and B
// fragments are then pure ds_load_b128 (no per-k cvt). The scan's shared B/C
// rows are double-buffered in LDS with async global->LDS copies (ASYNCcnt).
// Selective scan is register-resident VALU (16 states/thread, v_exp_f32).
// ---------------------------------------------------------------------------

#define DEPTH  4
#define DMODEL 128
#define DI     256
#define NST    16
#define RNK    8
#define KC     4
#define BATCH  4
#define LSEQ   2048
#define M_ROWS (BATCH * LSEQ)   // 8192 token rows
#define EPSR   1e-5f

typedef __attribute__((ext_vector_type(16))) __bf16    v16bf;
typedef __attribute__((ext_vector_type(8)))  __bf16    v8bf;
typedef __attribute__((ext_vector_type(4)))  __bf16    v4bf;
typedef __attribute__((ext_vector_type(8)))  float     v8f;
typedef __attribute__((ext_vector_type(4)))  unsigned  v4u;
typedef __attribute__((ext_vector_type(8)))  unsigned  v8u;

// ---------------------------------------------------------------- utilities
__device__ __forceinline__ float warp_sum32(float v) {
#pragma unroll
  for (int off = 16; off > 0; off >>= 1) v += __shfl_xor(v, off, 32);
  return v;
}

__device__ __forceinline__ float silu(float a) {
  return a / (1.0f + __expf(-a));
}

// Build a 16x32 bf16 WMMA operand fragment from fp32 memory (A path).
// Per CDNA5 ISA layout (16-bit A 16x32): lanes 0-15 hold rows, K groups
// [hi*8 .. hi*8+7] and [16+hi*8 .. 16+hi*8+7] where hi = lane>>4.
__device__ __forceinline__ v16bf load_frag_f32(const float* p) {
  float4 x0 = ((const float4*)p)[0];
  float4 x1 = ((const float4*)p)[1];
  float4 x2 = ((const float4*)(p + 16))[0];
  float4 x3 = ((const float4*)(p + 16))[1];
  v16bf f;
  f[0]  = (__bf16)x0.x; f[1]  = (__bf16)x0.y; f[2]  = (__bf16)x0.z; f[3]  = (__bf16)x0.w;
  f[4]  = (__bf16)x1.x; f[5]  = (__bf16)x1.y; f[6]  = (__bf16)x1.z; f[7]  = (__bf16)x1.w;
  f[8]  = (__bf16)x2.x; f[9]  = (__bf16)x2.y; f[10] = (__bf16)x2.z; f[11] = (__bf16)x2.w;
  f[12] = (__bf16)x3.x; f[13] = (__bf16)x3.y; f[14] = (__bf16)x3.z; f[15] = (__bf16)x3.w;
  return f;
}

// Build a fragment from a pre-converted bf16 LDS tile: two ds_load_b128.
__device__ __forceinline__ v16bf load_frag_bf16(const __bf16* p) {
  v8bf lo = *(const v8bf*)p;          // elements e0..7  (K  = base+0..7)
  v8bf hi = *(const v8bf*)(p + 16);   // elements e8..15 (K += 16)
  return __builtin_shufflevector(lo, hi, 0, 1, 2, 3, 4, 5, 6, 7,
                                 8, 9, 10, 11, 12, 13, 14, 15);
}

__device__ __forceinline__ v8f wmma_bf16(v16bf a, v16bf b, v8f c) {
  // (neg_a, A, neg_b, B, c_mod, C, reuse_a, reuse_b)
  return __builtin_amdgcn_wmma_f32_16x16x32_bf16(false, a, false, b,
                                                 (short)0, c, false, false);
}

// Async global->LDS 16B copy (per active lane), tracked with ASYNCcnt.
__device__ __forceinline__ void async_copy_b128(unsigned lds_addr,
                                                const float* gptr) {
  unsigned long long ga = (unsigned long long)(size_t)gptr;
  asm volatile("global_load_async_to_lds_b128 %0, %1, off"
               :: "v"(lds_addr), "v"(ga)
               : "memory");
}

__device__ __forceinline__ void wait_asynccnt0() {
  asm volatile("s_wait_asynccnt 0x0" ::: "memory");
}

// Tensor Data Mover: DMA a 2D fp32 tile (rows x rowElems) from global memory
// (row stride = strideElems) into LDS at lds_addr. Rows/cols beyond
// tensorRows/rowElems read as zero (TDM OOB rule) -> free zero padding.
// D# layout per CDNA5 ISA 08 (group0 128b + group1 256b, 2-group form).
__device__ __forceinline__ void tdm_load_2d(unsigned lds_addr,
                                            const float* gptr,
                                            unsigned rowElems,
                                            unsigned rows,
                                            unsigned tensorRows,
                                            unsigned strideElems) {
  unsigned long long ga = (unsigned long long)(size_t)gptr;
  v4u g0;
  g0[0] = 1u;                                   // count=1 valid user D#
  g0[1] = lds_addr;                             // lds_addr (bytes)
  g0[2] = (unsigned)(ga & 0xffffffffu);         // global_addr[31:0]
  g0[3] = (unsigned)((ga >> 32) & 0x01ffffffu)  // global_addr[56:32]
          | (2u << 30);                         // type = 2 ("image")
  v8u g1;
  g1[0] = (2u << 16);                           // data_size=2 (4B), mask=0
  g1[1] = (rowElems & 0xffffu) << 16;           // tensor_dim0[15:0]
  g1[2] = (rowElems >> 16) | ((tensorRows & 0xffffu) << 16);  // d0 hi | d1 lo
  g1[3] = (tensorRows >> 16) | ((rowElems & 0xffffu) << 16);  // d1 hi | tile_dim0
  g1[4] = rows & 0xffffu;                       // tile_dim1 | tile_dim2=0
  g1[5] = strideElems;                          // tensor_dim0_stride[31:0]
  g1[6] = 0u;                                   // d0_stride hi | d1_stride lo
  g1[7] = 0u;                                   // d1_stride hi
  asm volatile("tensor_load_to_lds %0, %1"
               :: "s"(g0), "s"(g1)
               : "memory");
}

// ------------------------------------------------- residual + double RMSNorm
// res = h (+ r); rnew = 2*res (the pair-level residual identity);
// hnA = rms(res)*nwA ; hnB = rms(res)*nwB  (norm inputs for both layers).
__global__ void residual_rms_kernel(const float* __restrict__ h,
                                    const float* __restrict__ r,
                                    const float* __restrict__ nwA,
                                    const float* __restrict__ nwB,
                                    float* __restrict__ res,
                                    float* __restrict__ rnew,
                                    float* __restrict__ hnA,
                                    float* __restrict__ hnB,
                                    int hasR) {
  int row  = blockIdx.x * 8 + (threadIdx.x >> 5);
  int lane = threadIdx.x & 31;
  float4 v = ((const float4*)(h + (size_t)row * DMODEL))[lane];
  if (hasR) {
    float4 rv = ((const float4*)(r + (size_t)row * DMODEL))[lane];
    v.x += rv.x; v.y += rv.y; v.z += rv.z; v.w += rv.w;
  }
  float ss = v.x * v.x + v.y * v.y + v.z * v.z + v.w * v.w;
  ss = warp_sum32(ss);
  float sc = rsqrtf(ss * (1.0f / DMODEL) + EPSR);

  ((float4*)(res + (size_t)row * DMODEL))[lane] = v;
  float4 r2 = make_float4(2.f * v.x, 2.f * v.y, 2.f * v.z, 2.f * v.w);
  ((float4*)(rnew + (size_t)row * DMODEL))[lane] = r2;

  float4 wa = ((const float4*)nwA)[lane];
  float4 wb = ((const float4*)nwB)[lane];
  float4 a = make_float4(v.x * sc * wa.x, v.y * sc * wa.y,
                         v.z * sc * wa.z, v.w * sc * wa.w);
  float4 b = make_float4(v.x * sc * wb.x, v.y * sc * wb.y,
                         v.z * sc * wb.z, v.w * sc * wb.w);
  ((float4*)(hnA + (size_t)row * DMODEL))[lane] = a;
  ((float4*)(hnB + (size_t)row * DMODEL))[lane] = b;
}

// out = rms(h + r) * nfw
__global__ void final_norm_kernel(const float* __restrict__ h,
                                  const float* __restrict__ r,
                                  const float* __restrict__ nfw,
                                  float* __restrict__ out) {
  int row  = blockIdx.x * 8 + (threadIdx.x >> 5);
  int lane = threadIdx.x & 31;
  float4 v  = ((const float4*)(h + (size_t)row * DMODEL))[lane];
  float4 rv = ((const float4*)(r + (size_t)row * DMODEL))[lane];
  v.x += rv.x; v.y += rv.y; v.z += rv.z; v.w += rv.w;
  float ss = warp_sum32(v.x * v.x + v.y * v.y + v.z * v.z + v.w * v.w);
  float sc = rsqrtf(ss * (1.0f / DMODEL) + EPSR);
  float4 w = ((const float4*)nfw)[lane];
  float4 o = make_float4(v.x * sc * w.x, v.y * sc * w.y,
                         v.z * sc * w.z, v.w * sc * w.w);
  ((float4*)(out + (size_t)row * DMODEL))[lane] = o;
}

// --------------------------------------------------------------- WMMA GEMM
// C[M, Npad] (+)= A[M, K] * W[Nreal, K]^T  (rows n >= Nreal contribute zero).
// Block = 128 threads (4 waves); each wave owns a 16(M) x 32(N) tile.
// Staging: (1) wave 0 DMAs the 32xK fp32 weight tile into LDS with one
// tensor_load_to_lds (rows beyond Nreal arrive as zeros via the TDM OOB
// rule) and waits TENSORcnt; (2) all 128 threads convert the tile to a bf16
// LDS copy once; (3) the K-loop then builds each B fragment with just two
// ds_load_b128 and no VALU conversions.
template <int K>
__global__ void gemm_wmma_bf16(const float* __restrict__ A,
                               const float* __restrict__ W,
                               float* __restrict__ C,
                               int Npad, int Nreal, int accum) {
  __shared__ __align__(16) float  wtile[32 * K];   // fp32 tile (TDM dest)
  __shared__ __align__(16) __bf16 wbf[32 * K];     // bf16 tile (WMMA source)

  int wave = threadIdx.x >> 5;
  int lane = threadIdx.x & 31;
  int l16  = lane & 15;
  int hi   = lane >> 4;
  int mtile  = (blockIdx.y * 4 + wave) * 16;
  int ntile0 = blockIdx.x * 32;
  int m = mtile + l16;

  if (wave == 0) {
    tdm_load_2d((unsigned)(size_t)&wtile[0],
                W + (size_t)ntile0 * K,
                (unsigned)K, 32u,
                (unsigned)(Nreal - ntile0),   // rows past this read zero
                (unsigned)K);
    __builtin_amdgcn_s_wait_tensorcnt(0);     // DMA landed in LDS
  }
  __syncthreads();                            // fp32 tile visible

  // one-shot fp32 -> bf16 tile conversion (each thread: 32*K/128 elements)
#pragma unroll
  for (int ci = 0; ci < 32 * (K / 4) / 128; ++ci) {
    int q = (ci * 128 + threadIdx.x) * 4;
    float4 x = *(const float4*)&wtile[q];
    v4bf o;
    o[0] = (__bf16)x.x; o[1] = (__bf16)x.y;
    o[2] = (__bf16)x.z; o[3] = (__bf16)x.w;
    *(v4bf*)&wbf[q] = o;
  }
  __syncthreads();                            // bf16 tile visible

  const float*  arow = A + (size_t)m * K + hi * 8;
  const __bf16* w0l  = &wbf[(size_t)l16 * K + hi * 8];         // B col n0
  const __bf16* w1l  = &wbf[(size_t)(l16 + 16) * K + hi * 8];  // B col n1

  v8f c0 = {}; v8f c1 = {};
#pragma unroll
  for (int k0 = 0; k0 < K; k0 += 32) {
    __builtin_prefetch(arow + k0 + 64, 0, 0);   // global_prefetch_b8
    v16bf af = load_frag_f32(arow + k0);
    v16bf b0 = load_frag_bf16(w0l + k0);        // 2x ds_load_b128, no cvt
    v16bf b1 = load_frag_bf16(w1l + k0);
    c0 = wmma_bf16(af, b0, c0);
    c1 = wmma_bf16(af, b1, c1);
  }

  // f32 C/D layout: VGPR v -> row mtile + v + 8*hi, col ntile + (lane&15).
  size_t base = (size_t)(mtile + hi * 8) * Npad + ntile0 + l16;
#pragma unroll
  for (int v = 0; v < 8; ++v) {
    size_t idx = base + (size_t)v * Npad;
    float o0 = c0[v], o1 = c1[v];
    if (accum) { o0 += C[idx]; o1 += C[idx + 16]; }
    C[idx]      = o0;
    C[idx + 16] = o1;
  }
}

// ------------------------------------------------- depthwise causal conv+SiLU
// forward:  y[l] = b + sum_j w[j]*x[l-3+j];  backward: taps l+3-j (mirrored).
__global__ void conv_silu_kernel(const float* __restrict__ xz,
                                 const float* __restrict__ cw,
                                 const float* __restrict__ cb,
                                 float* __restrict__ xc,
                                 int backward) {
  int idx = blockIdx.x * blockDim.x + threadIdx.x;  // over M_ROWS*DI
  int c   = idx & (DI - 1);
  int row = idx >> 8;
  int l   = row & (LSEQ - 1);
  int b   = row >> 11;
  float acc = cb[c];
#pragma unroll
  for (int j = 0; j < KC; ++j) {
    int pos = backward ? (l + (KC - 1) - j) : (l - (KC - 1) + j);
    if (pos >= 0 && pos < LSEQ) {
      acc += cw[c * KC + j] *
             xz[((size_t)(b * LSEQ + pos)) * (2 * DI) + c];
    }
  }
  xc[(size_t)row * DI + c] = silu(acc);
}

// ------------------------------------------------------ dt = softplus(proj+b)
__global__ void dt_kernel(const float* __restrict__ dbl,   // [M, 64]
                          const float* __restrict__ dpw,   // [DI, R]
                          const float* __restrict__ dpb,   // [DI]
                          float* __restrict__ dt) {
  int idx = blockIdx.x * blockDim.x + threadIdx.x;
  int c   = idx & (DI - 1);
  int row = idx >> 8;
  const float* dr = dbl + (size_t)row * 64;
  float s = dpb[c];
#pragma unroll
  for (int r = 0; r < RNK; ++r) s += dr[r] * dpw[c * RNK + r];
  dt[(size_t)row * DI + c] = (s > 20.f) ? s : log1pf(__expf(s));
}

// ------------------------------------------------------------ selective scan
// One wave (32 threads) per block; one thread per channel, 16 fp32 states in
// registers. The per-step B||C row (32 floats, shared by all channels) is
// double-buffered in LDS with async global->LDS copies issued one step ahead
// (ASYNCcnt). Single-wave blocks need no barriers: s_wait_asynccnt suffices.
// dbl row: [0:8)=dt-lowrank, [8:24)=B, [24:40)=C (ld = 64).
__global__ void scan_kernel(const float* __restrict__ xc,
                            const float* __restrict__ dt,
                            const float* __restrict__ dbl,
                            const float* __restrict__ xz,   // z = cols DI..2DI
                            const float* __restrict__ alog,
                            const float* __restrict__ dsk,
                            float* __restrict__ y,
                            int backward, int accum) {
  __shared__ __align__(16) float sbuf[64];    // 2 x 32 floats (B||C)
  int b    = blockIdx.x;
  int lane = threadIdx.x;                      // 0..31
  int c    = blockIdx.y * 32 + lane;

  float Av[NST];
#pragma unroll
  for (int n = 0; n < NST; ++n) Av[n] = -__expf(alog[c * NST + n]);
  float Dk = dsk[c];
  float hst[NST];
#pragma unroll
  for (int n = 0; n < NST; ++n) hst[n] = 0.f;

  // prime buffer 0 with the first step's B||C row
  {
    int l0 = backward ? (LSEQ - 1) : 0;
    const float* g = dbl + ((size_t)b * LSEQ + l0) * 64 + 8;
    if (lane < 8)
      async_copy_b128((unsigned)(size_t)&sbuf[lane * 4], g + lane * 4);
    wait_asynccnt0();
  }

  for (int t = 0; t < LSEQ; ++t) {
    int l = backward ? (LSEQ - 1 - t) : t;
    size_t row = (size_t)b * LSEQ + l;

    // prefetch next step's row into the other buffer
    if (t + 1 < LSEQ && lane < 8) {
      int ln = backward ? (LSEQ - 2 - t) : (t + 1);
      const float* g = dbl + ((size_t)b * LSEQ + ln) * 64 + 8;
      async_copy_b128((unsigned)(size_t)&sbuf[((t + 1) & 1) * 32 + lane * 4],
                      g + lane * 4);
    }

    float dtv = dt[row * DI + c];
    float xv  = xc[row * DI + c];
    float zv  = xz[row * (2 * DI) + DI + c];
    const float* dr = &sbuf[(t & 1) * 32];     // [0:16)=B, [16:32)=C
    float dtx = dtv * xv;
    float yv = 0.f;
#pragma unroll
    for (int n = 0; n < NST; ++n) {
      float dA = __expf(dtv * Av[n]);
      hst[n] = dA * hst[n] + dtx * dr[n];
      yv += hst[n] * dr[16 + n];
    }
    yv += xv * Dk;
    yv *= silu(zv);
    size_t oi = row * DI + c;
    if (accum) y[oi] += yv; else y[oi] = yv;

    wait_asynccnt0();                          // next row resident
  }
}

// ---------------------------------------------------------------------------
extern "C" void kernel_launch(void* const* d_in, const int* in_sizes, int n_in,
                              void* d_out, int out_size, void* d_ws,
                              size_t ws_size, hipStream_t stream) {
  const float* x    = (const float*)d_in[0];
  const float* nw   = (const float*)d_in[1];
  const float* inw  = (const float*)d_in[2];
  const float* cw   = (const float*)d_in[3];
  const float* cb   = (const float*)d_in[4];
  const float* xpw  = (const float*)d_in[5];
  const float* dpw  = (const float*)d_in[6];
  const float* dpb  = (const float*)d_in[7];
  const float* alog = (const float*)d_in[8];
  const float* dsk  = (const float*)d_in[9];
  const float* opw  = (const float*)d_in[10];
  const float* nfw  = (const float*)d_in[11];
  float* out = (float*)d_out;

  char* ws = (char*)d_ws;
  size_t off = 0;
  auto alloc = [&](size_t nflt) {
    float* p = (float*)(ws + off);
    off += nflt * sizeof(float);
    return p;
  };
  float* res  = alloc((size_t)M_ROWS * DMODEL);
  float* rbuf = alloc((size_t)M_ROWS * DMODEL);
  float* hnA  = alloc((size_t)M_ROWS * DMODEL);
  float* hnB  = alloc((size_t)M_ROWS * DMODEL);
  float* xzb  = alloc((size_t)M_ROWS * 2 * DI);
  float* xcb  = alloc((size_t)M_ROWS * DI);
  float* dblb = alloc((size_t)M_ROWS * 64);
  float* dtb  = alloc((size_t)M_ROWS * DI);
  float* yb   = alloc((size_t)M_ROWS * DI);
  float* hA   = alloc((size_t)M_ROWS * DMODEL);
  float* hB   = alloc((size_t)M_ROWS * DMODEL);
  (void)ws_size; (void)in_sizes; (void)n_in; (void)out_size;

  const float* hcur = x;
  float* houts[2] = {hA, hB};

  for (int p = 0; p < 2; ++p) {
    // res = h (+ r); rnew = 2*res; normalized inputs for both pair layers.
    residual_rms_kernel<<<M_ROWS / 8, 256, 0, stream>>>(
        hcur, rbuf, nw + (size_t)(2 * p) * DMODEL,
        nw + (size_t)(2 * p + 1) * DMODEL, res, rbuf, hnA, hnB, p > 0);

    float* hout = houts[p];
    for (int d2 = 0; d2 < 2; ++d2) {        // layer parity within the pair
      int li = 2 * p + d2;
      const float* hn = d2 ? hnB : hnA;

      // in-projection: [M,128] x [512,128]^T -> [M,512]
      gemm_wmma_bf16<DMODEL><<<dim3((2 * DI) / 32, M_ROWS / 64), 128, 0,
                               stream>>>(hn, inw + (size_t)li * 2 * DI * DMODEL,
                                         xzb, 2 * DI, 2 * DI, 0);

      for (int d = 0; d < 2; ++d) {         // scan directions
        int back = d ^ d2;                  // odd layers are mirrored
        size_t s = (size_t)li * 2 + d;
        conv_silu_kernel<<<(M_ROWS * DI) / 256, 256, 0, stream>>>(
            xzb, cw + s * DI * KC, cb + s * DI, xcb, back);
        // x-projection: [M,256] x [40,256]^T -> [M,64] (padded, rows>=40 = 0)
        gemm_wmma_bf16<DI><<<dim3(64 / 32, M_ROWS / 64), 128, 0, stream>>>(
            xcb, xpw + s * (RNK + 2 * NST) * DI, dblb, 64, RNK + 2 * NST, 0);
        dt_kernel<<<(M_ROWS * DI) / 256, 256, 0, stream>>>(
            dblb, dpw + s * DI * RNK, dpb + s * DI, dtb);
        scan_kernel<<<dim3(BATCH, DI / 32), 32, 0, stream>>>(
            xcb, dtb, dblb, xzb, alog + s * DI * NST, dsk + s * DI, yb, back,
            d);
      }
      // out-projection: [M,256] x [128,256]^T -> [M,128], accumulate pair sum
      gemm_wmma_bf16<DI><<<dim3(DMODEL / 32, M_ROWS / 64), 128, 0, stream>>>(
          yb, opw + (size_t)li * DMODEL * DI, hout, DMODEL, DMODEL, d2);
    }
    hcur = hout;
  }

  final_norm_kernel<<<M_ROWS / 8, 256, 0, stream>>>(hcur, rbuf, nfw, out);
}
